// StreamMamba_764504179278
// MI455X (gfx1250) — matrix operand order repaired and checked by reading, not compiled, not run on hardware
//
#include <hip/hip_runtime.h>
#include <hip/hip_bf16.h>

typedef __bf16 bf16;
typedef __attribute__((ext_vector_type(16))) __bf16 v16bf;
typedef __attribute__((ext_vector_type(8)))  __bf16 bf16x8;
typedef __attribute__((ext_vector_type(8)))  float  v8f;

// Problem constants (reference file)
static constexpr int Bc  = 4;
static constexpr int Tc  = 2048;
static constexpr int DM  = 768;
static constexpr int LYR = 2;
static constexpr int DI  = 1536;
static constexpr int Nst = 64;
static constexpr int DC  = 4;
static constexpr int Rr  = 48;
static constexpr int BT  = Bc * Tc;          // 8192 rows
static constexpr int PN  = Rr + 2 * Nst;     // 176 = dt(48) | B(64) | C(64)

__device__ __forceinline__ v16bf cat8(bf16x8 lo, bf16x8 hi) {
  return __builtin_shufflevector(lo, hi, 0,1,2,3,4,5,6,7,8,9,10,11,12,13,14,15);
}

__device__ __forceinline__ float silu(float x) {
  return x / (1.0f + __expf(-x));
}

// ---------------------------------------------------------------------------
// fp32 -> bf16 bulk convert (weights)
// ---------------------------------------------------------------------------
__global__ void f2bf_kernel(const float* __restrict__ in, bf16* __restrict__ out, int n) {
  int i = blockIdx.x * 256 + threadIdx.x;
  if (i < n) out[i] = (bf16)in[i];
}

// ---------------------------------------------------------------------------
// x = x * mask ; produce fp32 + bf16 copies
// ---------------------------------------------------------------------------
__global__ void mask_kernel(const float* __restrict__ x, const float* __restrict__ mask,
                            float* __restrict__ xf, bf16* __restrict__ xbf) {
  int idx = blockIdx.x * 256 + threadIdx.x;      // over BT*DM
  int row = idx / DM;
  float v = x[idx] * mask[row];
  xf[idx]  = v;
  xbf[idx] = (bf16)v;
}

// ---------------------------------------------------------------------------
// C[M x N] (fp32) = A[M x K] (bf16, row major) * W[N x K]^T (bf16, row major)
// One wave32 computes a 64 x (16*NSUB) output tile: 4 x NSUB grid of 16x16
// WMMA sub-tiles. Tile shape is compile-time -> fully straight-line body
// (no EXEC manipulation around WMMA, no divergence, no spill-inducing guards).
// Per K-step of 32 (NSUB=4): 8 A-frag + 8 B-frag b128 loads feed 16 WMMAs.
// ---------------------------------------------------------------------------
template <int NSUB>
__global__ void gemm_wmma_bf16_tile(const bf16* __restrict__ A, const bf16* __restrict__ W,
                                    float* __restrict__ C, int M, int N, int K,
                                    int col_base, int ntiles) {
  int wave = (blockIdx.x * blockDim.x + threadIdx.x) >> 5;
  int lane = threadIdx.x & 31;
  int mt = wave / ntiles;
  int nt = wave - mt * ntiles;
  if (mt >= (M >> 6)) return;            // wave-uniform

  int rowb   = (mt << 6) + (lane & 15);            // A row base (sub-tile i adds 16*i)
  int colb   = col_base + nt * (16 * NSUB) + (lane & 15);  // B col base (sub-tile j adds 16*j)
  int a_koff = (lane >> 4) << 3;         // 0 or 8  (A 16x32: K 0..7/16..23 vs 8..15/24..31)
  int b_koff = (lane >> 4) << 3;         // 0 or 8  (B 32x16: K 0..7/16..23 vs 8..15/24..31)

  const bf16* Ar = A + (size_t)rowb * K;

  v8f zero = {};
  v8f acc[4][NSUB];
  #pragma unroll
  for (int i = 0; i < 4; ++i)
    #pragma unroll
    for (int j = 0; j < NSUB; ++j) acc[i][j] = zero;

  for (int k0 = 0; k0 < K; k0 += 32) {
    v16bf af[4], bfr[NSUB];
    #pragma unroll
    for (int i = 0; i < 4; ++i) {
      const bf16* Ap = Ar + (size_t)(i << 4) * K + k0 + a_koff;
      af[i] = cat8(*(const bf16x8*)Ap, *(const bf16x8*)(Ap + 16));
    }
    #pragma unroll
    for (int j = 0; j < NSUB; ++j) {
      const bf16* Bp = W + (size_t)(colb + (j << 4)) * K + k0 + b_koff;
      bfr[j] = cat8(*(const bf16x8*)Bp, *(const bf16x8*)(Bp + 16));
    }
    #pragma unroll
    for (int i = 0; i < 4; ++i) {
      #pragma unroll
      for (int j = 0; j < NSUB; ++j) {
        acc[i][j] = __builtin_amdgcn_wmma_f32_16x16x32_bf16(
            false, af[i], false, bfr[j], (short)0, acc[i][j], false, false);
      }
    }
  }

  // Store: C/D layout -> VGPR r holds row (8*(lane>=16) + r), col = lane&15
  #pragma unroll
  for (int i = 0; i < 4; ++i) {
    int rbase = (mt << 6) + (i << 4) + ((lane >> 4) << 3);
    #pragma unroll
    for (int j = 0; j < NSUB; ++j) {
      float* Cp = C + (size_t)rbase * N + (colb + (j << 4));
      #pragma unroll
      for (int r = 0; r < 8; ++r) Cp[(size_t)r * N] = acc[i][j][r];
    }
  }
}

// ---------------------------------------------------------------------------
// Depthwise causal conv (width 4) over u = xz[..., :DI], + bias, SiLU
// ---------------------------------------------------------------------------
__global__ void conv_silu_kernel(const float* __restrict__ xz,
                                 const float* __restrict__ wc,  // DI x DC
                                 const float* __restrict__ bc,  // DI
                                 float* __restrict__ ucf, bf16* __restrict__ ucbf) {
  int idx = blockIdx.x * 256 + threadIdx.x;      // over BT*DI
  int d = idx % DI;
  int row = idx / DI;
  int t = row % Tc;
  const float* w = wc + d * DC;
  float acc = bc[d];
  #pragma unroll
  for (int j = 0; j < DC; ++j) {
    int tt = t - (DC - 1) + j;
    if (tt >= 0) acc += w[j] * xz[(size_t)(row - t + tt) * (2 * DI) + d];
  }
  float s = silu(acc);
  ucf[idx]  = s;
  ucbf[idx] = (bf16)s;
}

// ---------------------------------------------------------------------------
// delta = softplus(dt @ W_dt^T + b_dt) ; dt = proj[row][0:48]
// ---------------------------------------------------------------------------
__global__ void delta_kernel(const float* __restrict__ proj,
                             const float* __restrict__ wdt,  // DI x R
                             const float* __restrict__ bdt,  // DI
                             float* __restrict__ delta) {
  __shared__ float dt[Rr];
  int row = blockIdx.x / (DI / 256);
  int d   = (blockIdx.x % (DI / 256)) * 256 + threadIdx.x;
  if (threadIdx.x < Rr) dt[threadIdx.x] = proj[(size_t)row * PN + threadIdx.x];
  __syncthreads();
  const float* w = wdt + (size_t)d * Rr;
  float acc = bdt[d];
  #pragma unroll 8
  for (int r = 0; r < Rr; ++r) acc += dt[r] * w[r];
  float sp = (acc > 20.0f) ? acc : log1pf(__expf(acc));
  delta[(size_t)row * DI + d] = sp;
}

// ---------------------------------------------------------------------------
// Selective scan: one wave32 per (b, d). Lane holds states n = lane, lane+32.
// h = h*exp(delta*A) + delta*B_t*u ; y = sum_n h*C_t  (shfl_xor reduction)
// ---------------------------------------------------------------------------
__global__ void scan_kernel(const float* __restrict__ delta,
                            const float* __restrict__ ucf,
                            const float* __restrict__ proj,
                            const float* __restrict__ alog,  // DI x N
                            float* __restrict__ y) {
  int wave = (blockIdx.x * blockDim.x + threadIdx.x) >> 5;
  int lane = threadIdx.x & 31;
  int b = wave / DI;
  int d = wave - b * DI;

  float A0 = -__expf(alog[(size_t)d * Nst + lane]);
  float A1 = -__expf(alog[(size_t)d * Nst + lane + 32]);
  float h0 = 0.0f, h1 = 0.0f;

  size_t rowbase = (size_t)b * Tc;
  for (int t = 0; t < Tc; ++t) {
    size_t row = rowbase + t;
    float dlt = delta[row * DI + d];
    float u   = ucf[row * DI + d];
    const float* pr = proj + row * PN;
    float Bv0 = pr[Rr + lane],        Bv1 = pr[Rr + lane + 32];
    float C0  = pr[Rr + Nst + lane],  C1  = pr[Rr + Nst + lane + 32];

    h0 = h0 * __expf(dlt * A0) + dlt * Bv0 * u;
    h1 = h1 * __expf(dlt * A1) + dlt * Bv1 * u;

    float p = h0 * C0 + h1 * C1;
    p += __shfl_xor(p, 16, 32);
    p += __shfl_xor(p,  8, 32);
    p += __shfl_xor(p,  4, 32);
    p += __shfl_xor(p,  2, 32);
    p += __shfl_xor(p,  1, 32);
    if (lane == 0) y[row * DI + d] = p;
  }
}

// ---------------------------------------------------------------------------
// y = (y + u_c * D) * silu(z) -> bf16 for out_proj GEMM
// ---------------------------------------------------------------------------
__global__ void gate_kernel(const float* __restrict__ y, const float* __restrict__ ucf,
                            const float* __restrict__ Dp, const float* __restrict__ xz,
                            bf16* __restrict__ ygbf) {
  int idx = blockIdx.x * 256 + threadIdx.x;      // over BT*DI
  int d = idx % DI;
  size_t row = (size_t)(idx / DI);
  float z  = xz[row * (2 * DI) + DI + d];
  float yy = (y[idx] + ucf[idx] * Dp[d]) * silu(z);
  ygbf[idx] = (bf16)yy;
}

// ---------------------------------------------------------------------------
// LayerNorm + FiLM + residual. One block per row of 768.
// ---------------------------------------------------------------------------
__global__ void ln_film_res_kernel(const float* __restrict__ h, const float* __restrict__ resid,
                                   const float* __restrict__ gamma, const float* __restrict__ beta,
                                   const float* __restrict__ lnw, const float* __restrict__ lnb,
                                   float* __restrict__ xout, bf16* __restrict__ xout_bf) {
  __shared__ float s1[256];
  __shared__ float s2[256];
  int row = blockIdx.x;
  int tid = threadIdx.x;
  const float* hr = h + (size_t)row * DM;
  float a0 = hr[tid], a1 = hr[tid + 256], a2 = hr[tid + 512];
  s1[tid] = a0 + a1 + a2;
  s2[tid] = a0 * a0 + a1 * a1 + a2 * a2;
  __syncthreads();
  for (int o = 128; o > 0; o >>= 1) {
    if (tid < o) { s1[tid] += s1[tid + o]; s2[tid] += s2[tid + o]; }
    __syncthreads();
  }
  float mean = s1[0] * (1.0f / DM);
  float var  = s2[0] * (1.0f / DM) - mean * mean;
  float inv  = rsqrtf(var + 1e-5f);
  float vals[3] = {a0, a1, a2};
  #pragma unroll
  for (int j = 0; j < 3; ++j) {
    int i = tid + j * 256;
    size_t gi = (size_t)row * DM + i;
    float hn = (vals[j] - mean) * inv * lnw[i] + lnb[i];
    float o  = gamma[gi] * hn + beta[gi] + resid[gi];
    xout[gi]    = o;
    xout_bf[gi] = (bf16)o;
  }
}

// ---------------------------------------------------------------------------
extern "C" void kernel_launch(void* const* d_in, const int* in_sizes, int n_in,
                              void* d_out, int out_size, void* d_ws, size_t ws_size,
                              hipStream_t stream) {
  (void)in_sizes; (void)n_in; (void)out_size; (void)ws_size;

  const float* x      = (const float*)d_in[0];
  const float* mask   = (const float*)d_in[1];
  const float* fgam   = (const float*)d_in[2];
  const float* fbet   = (const float*)d_in[3];
  const float* w_in   = (const float*)d_in[4];   // LYR x 2DI x DM
  const float* convw  = (const float*)d_in[5];   // LYR x DI x DC
  const float* convb  = (const float*)d_in[6];   // LYR x DI
  const float* w_xp   = (const float*)d_in[7];   // LYR x PN x DI
  const float* w_dt   = (const float*)d_in[8];   // LYR x DI x R
  const float* b_dt   = (const float*)d_in[9];   // LYR x DI
  const float* alog   = (const float*)d_in[10];  // LYR x DI x N
  const float* Dp     = (const float*)d_in[11];  // LYR x DI
  const float* w_out  = (const float*)d_in[12];  // LYR x DM x DI
  const float* lnw    = (const float*)d_in[13];
  const float* lnb    = (const float*)d_in[14];

  // ---- carve workspace (256B aligned regions) ----
  char* p = (char*)d_ws;
  auto carve = [&](size_t bytes) { char* r = p; p += (bytes + 255) & ~(size_t)255; return (void*)r; };
  float* xf   = (float*)carve((size_t)BT * DM * 4);
  bf16*  xbf  = (bf16*) carve((size_t)BT * DM * 2);
  bf16*  winb = (bf16*) carve((size_t)LYR * 2 * DI * DM * 2);
  bf16*  wxpb = (bf16*) carve((size_t)LYR * PN * DI * 2);
  bf16*  wotb = (bf16*) carve((size_t)LYR * DM * DI * 2);
  float* xz   = (float*)carve((size_t)BT * 2 * DI * 4);
  float* ucf  = (float*)carve((size_t)BT * DI * 4);
  bf16*  ucbf = (bf16*) carve((size_t)BT * DI * 2);
  float* proj = (float*)carve((size_t)BT * PN * 4);
  float* delt = (float*)carve((size_t)BT * DI * 4);
  float* ybuf = (float*)carve((size_t)BT * DI * 4);
  bf16*  ygbf = (bf16*) carve((size_t)BT * DI * 2);
  float* mo   = (float*)carve((size_t)BT * DM * 4);

  auto cvt = [&](const float* src, bf16* dst, int n) {
    f2bf_kernel<<<(n + 255) / 256, 256, 0, stream>>>(src, dst, n);
  };
  // N = full 64-wide tiles (NSUB=4) + optional 16/32/48-wide remainder tile.
  auto gemm = [&](const bf16* A, const bf16* Wb, float* C, int M, int N, int K) {
    int full = N / 64;
    if (full > 0) {
      int waves = (M / 64) * full;
      gemm_wmma_bf16_tile<4><<<(waves * 32 + 255) / 256, 256, 0, stream>>>(
          A, Wb, C, M, N, K, 0, full);
    }
    int rem = (N - full * 64) / 16;
    int rwaves = (M / 64);
    int rblocks = (rwaves * 32 + 255) / 256;
    if (rem == 3) {
      gemm_wmma_bf16_tile<3><<<rblocks, 256, 0, stream>>>(A, Wb, C, M, N, K, full * 64, 1);
    } else if (rem == 2) {
      gemm_wmma_bf16_tile<2><<<rblocks, 256, 0, stream>>>(A, Wb, C, M, N, K, full * 64, 1);
    } else if (rem == 1) {
      gemm_wmma_bf16_tile<1><<<rblocks, 256, 0, stream>>>(A, Wb, C, M, N, K, full * 64, 1);
    }
  };

  // ---- convert weights to bf16 (all layers at once) ----
  cvt(w_in,  winb, LYR * 2 * DI * DM);
  cvt(w_xp,  wxpb, LYR * PN * DI);
  cvt(w_out, wotb, LYR * DM * DI);

  // ---- x = x * mask ----
  mask_kernel<<<(BT * DM) / 256, 256, 0, stream>>>(x, mask, xf, xbf);

  for (int L = 0; L < LYR; ++L) {
    // in_proj: xz = x @ W_in^T   (8192 x 3072)
    gemm(xbf, winb + (size_t)L * 2 * DI * DM, xz, BT, 2 * DI, DM);

    // depthwise causal conv + SiLU on u
    conv_silu_kernel<<<(BT * DI) / 256, 256, 0, stream>>>(
        xz, convw + (size_t)L * DI * DC, convb + (size_t)L * DI, ucf, ucbf);

    // x_proj: proj = u_c @ W_x^T  (8192 x 176 = 2 full tiles + 48-wide remainder)
    gemm(ucbf, wxpb + (size_t)L * PN * DI, proj, BT, PN, DI);

    // delta = softplus(dt @ W_dt^T + b_dt)
    delta_kernel<<<BT * (DI / 256), 256, 0, stream>>>(
        proj, w_dt + (size_t)L * DI * Rr, b_dt + (size_t)L * DI, delt);

    // sequential selective scan (wave per (b,d))
    scan_kernel<<<(Bc * DI * 32) / 256, 256, 0, stream>>>(
        delt, ucf, proj, alog + (size_t)L * DI * Nst, ybuf);

    // gate: y = (y + u_c*D) * silu(z)
    gate_kernel<<<(BT * DI) / 256, 256, 0, stream>>>(
        ybuf, ucf, Dp + (size_t)L * DI, xz, ygbf);

    // out_proj: mo = y @ W_out^T  (8192 x 768)
    gemm(ygbf, wotb + (size_t)L * DM * DI, mo, BT, DM, DI);

    // layernorm + FiLM + residual
    float* xo = (L == LYR - 1) ? (float*)d_out : xf;
    ln_film_res_kernel<<<BT, 256, 0, stream>>>(
        mo, xf, fgam + (size_t)L * BT * DM, fbet + (size_t)L * BT * DM,
        lnw + (size_t)L * DM, lnb + (size_t)L * DM, xo, xbf);
  }
}
